// ScaledDotProductAttention_46196668236308
// MI455X (gfx1250) — matrix operand order repaired and checked by reading, not compiled, not run on hardware
//
#include <hip/hip_runtime.h>

typedef __attribute__((ext_vector_type(16))) _Float16 v16h;
typedef __attribute__((ext_vector_type(8)))  _Float16 v8h;
typedef __attribute__((ext_vector_type(2)))  _Float16 v2h;
typedef __attribute__((ext_vector_type(8)))  float    v8f;
typedef __attribute__((ext_vector_type(4)))  float    v4f;

#define SEQ   8192
#define DIM   256
#define TK    64            // keys per iteration (four 16-col S tiles)
#define WAVES 4
#define BLOCK (WAVES * 32)  // wave32
#define QROWS_PER_BLOCK (WAVES * 16)

__global__ __launch_bounds__(BLOCK) void fa_wmma_kernel(
    const float* __restrict__ Q,
    const float* __restrict__ Kmat,
    const float* __restrict__ V,
    float* __restrict__ O)
{
    // LDS staging: K row-major (f16), V transposed (f16), per-wave P buffer
    __shared__ __align__(32) _Float16 sK [TK][DIM];       // 32 KB
    __shared__ __align__(32) _Float16 sVt[DIM][TK];       // 32 KB
    __shared__ __align__(32) _Float16 sP [WAVES][16][TK]; // 8 KB

    const int tid   = threadIdx.x;
    const int wave  = tid >> 5;
    const int lane  = tid & 31;
    const int lhalf = lane >> 4;   // 0: lanes 0-15, 1: lanes 16-31
    const int l16   = lane & 15;

    const int qbase = blockIdx.x * QROWS_PER_BLOCK + wave * 16;
    // 1/sqrt(256) * log2(e): scores land in log2 domain -> softmax via v_exp_f32
    const float qscale = 0.0625f * 1.44269504088896340736f;

    // ---- Load Q tile into A-matrix fragments (16x32 f16 per chunk), scale folded in.
    // A k order used here (contiguous, same as B): lanes 0-15 -> row M=lane, K=0..15;
    //                           lanes 16-31 -> row M=lane-16, K=16..31
    v16h aq[8];
    {
        const float* qrow = Q + (size_t)(qbase + l16) * DIM;
#pragma unroll
        for (int c = 0; c < 8; ++c) {
            const int k0 = 32 * c + (lhalf ? 16 : 0);
            const int k1 = k0 + 8;
            v4f qa = *(const v4f*)(qrow + k0);
            v4f qb = *(const v4f*)(qrow + k0 + 4);
            v4f qc = *(const v4f*)(qrow + k1);
            v4f qd = *(const v4f*)(qrow + k1 + 4);
#pragma unroll
            for (int i = 0; i < 4; ++i) {
                aq[c][i]      = (_Float16)(qa[i] * qscale);
                aq[c][4 + i]  = (_Float16)(qb[i] * qscale);
                aq[c][8 + i]  = (_Float16)(qc[i] * qscale);
                aq[c][12 + i] = (_Float16)(qd[i] * qscale);
            }
        }
    }

    // ---- Online-softmax state (C layout: reg r, lane -> row m = r + 8*lhalf, col = l16)
    float mrow[8], lrow[8];
    v8f acc[16];  // 16 output chunks of 16 cols -> full 16x256 O tile
#pragma unroll
    for (int r = 0; r < 8; ++r) { mrow[r] = -1.0e30f; lrow[r] = 0.0f; }
#pragma unroll
    for (int ch = 0; ch < 16; ++ch)
#pragma unroll
        for (int r = 0; r < 8; ++r) acc[ch][r] = 0.0f;

    for (int kb = 0; kb < SEQ; kb += TK) {
        __syncthreads();  // protect LDS from previous iteration's readers

        // ---- Cooperative stage (row pairs): K -> sK (row-major f16, packed b64
        // stores), V -> sVt (transposed f16, packed 2-row b32 stores)
        for (int e = tid * 4; e < (TK / 2) * DIM; e += BLOCK * 4) {
            const int rp  = e >> 8;        // row-pair index: rows 2rp, 2rp+1
            const int col = e & (DIM - 1);
            const size_t g0 = (size_t)(kb + 2 * rp) * DIM + col;
            const size_t g1 = (size_t)(kb + 2 * rp + 1) * DIM + col;
            v4f ka = *(const v4f*)(Kmat + g0);
            v4f kc = *(const v4f*)(Kmat + g1);
            v4f va = *(const v4f*)(V + g0);
            v4f vb = *(const v4f*)(V + g1);
#pragma unroll
            for (int j = 0; j < 4; ++j) {
                sK[2 * rp][col + j]     = (_Float16)ka[j];
                sK[2 * rp + 1][col + j] = (_Float16)kc[j];
                v2h pv; pv[0] = (_Float16)va[j]; pv[1] = (_Float16)vb[j];
                *(v2h*)&sVt[col + j][2 * rp] = pv;   // packed cvt + b32 store
            }
        }
        __syncthreads();

        // ---- S = Q * K^T for 64 keys: four 16x16 C tiles, K-dim = 256 = 8 chunks.
        // B fragments double-buffered in registers: issue chunk c+1 loads before
        // chunk c WMMAs so waits are dscnt<=N instead of 0.
        v8f s0, s1, s2, s3;
#pragma unroll
        for (int r = 0; r < 8; ++r) { s0[r] = 0.f; s1[r] = 0.f; s2[r] = 0.f; s3[r] = 0.f; }
        v16h bk0, bk1, bk2, bk3;
        {
            const int dco = lhalf * 16;
            bk0 = *(const v16h*)&sK[l16][dco];
            bk1 = *(const v16h*)&sK[16 + l16][dco];
            bk2 = *(const v16h*)&sK[32 + l16][dco];
            bk3 = *(const v16h*)&sK[48 + l16][dco];
        }
#pragma unroll
        for (int c = 0; c < 8; ++c) {
            v16h nb0, nb1, nb2, nb3;
            if (c < 7) {
                const int dco = 32 * (c + 1) + lhalf * 16;
                nb0 = *(const v16h*)&sK[l16][dco];
                nb1 = *(const v16h*)&sK[16 + l16][dco];
                nb2 = *(const v16h*)&sK[32 + l16][dco];
                nb3 = *(const v16h*)&sK[48 + l16][dco];
            }
            s0 = __builtin_amdgcn_wmma_f32_16x16x32_f16(false, aq[c], false, bk0,
                                                        (short)0, s0, false, false);
            s1 = __builtin_amdgcn_wmma_f32_16x16x32_f16(false, aq[c], false, bk1,
                                                        (short)0, s1, false, false);
            s2 = __builtin_amdgcn_wmma_f32_16x16x32_f16(false, aq[c], false, bk2,
                                                        (short)0, s2, false, false);
            s3 = __builtin_amdgcn_wmma_f32_16x16x32_f16(false, aq[c], false, bk3,
                                                        (short)0, s3, false, false);
            if (c < 7) { bk0 = nb0; bk1 = nb1; bk2 = nb2; bk3 = nb3; }
        }

        // ---- Online softmax update in log2 domain (row = r + 8*lhalf across 16 lanes)
        float alpha[8];
#pragma unroll
        for (int r = 0; r < 8; ++r) {
            float tm = fmaxf(fmaxf(s0[r], s1[r]), fmaxf(s2[r], s3[r]));
#pragma unroll
            for (int m = 8; m >= 1; m >>= 1) tm = fmaxf(tm, __shfl_xor(tm, m, 32));
            const float mn = fmaxf(mrow[r], tm);
            alpha[r] = __builtin_amdgcn_exp2f(mrow[r] - mn);
            const float p0 = __builtin_amdgcn_exp2f(s0[r] - mn);
            const float p1 = __builtin_amdgcn_exp2f(s1[r] - mn);
            const float p2 = __builtin_amdgcn_exp2f(s2[r] - mn);
            const float p3 = __builtin_amdgcn_exp2f(s3[r] - mn);
            float ps = (p0 + p1) + (p2 + p3);
#pragma unroll
            for (int m = 8; m >= 1; m >>= 1) ps += __shfl_xor(ps, m, 32);
            lrow[r] = lrow[r] * alpha[r] + ps;
            mrow[r] = mn;
            // stash P (f16) for re-layout; row m = r + 8*lhalf
            const int m = r + 8 * lhalf;
            sP[wave][m][l16]      = (_Float16)p0;
            sP[wave][m][16 + l16] = (_Float16)p1;
            sP[wave][m][32 + l16] = (_Float16)p2;
            sP[wave][m][48 + l16] = (_Float16)p3;
        }

        // ---- Issue P A-fragment reloads now; the rescale VALU below hides the
        // DS latency (same-wave DS ops are in-order, so data is coherent).
        v8h pa = *(const v8h*)&sP[wave][l16][lhalf ? 16 : 0];
        v8h pb = *(const v8h*)&sP[wave][l16][lhalf ? 24 : 8];
        v8h pc = *(const v8h*)&sP[wave][l16][32 + (lhalf ? 16 : 0)];
        v8h pd = *(const v8h*)&sP[wave][l16][32 + (lhalf ? 24 : 8)];

        // ---- Rescale accumulators by alpha (per row)
#pragma unroll
        for (int ch = 0; ch < 16; ++ch)
#pragma unroll
            for (int r = 0; r < 8; ++r) acc[ch][r] *= alpha[r];

        v16h ap0, ap1;
#pragma unroll
        for (int i = 0; i < 8; ++i) {
            ap0[i] = pa[i]; ap0[8 + i] = pb[i];
            ap1[i] = pc[i]; ap1[8 + i] = pd[i];
        }

        // ---- O += P * V : two WMMAs per 16-col output chunk, V fragments
        // double-buffered one chunk ahead.
        v16h cv0 = *(const v16h*)&sVt[l16][lhalf * 16];
        v16h cv1 = *(const v16h*)&sVt[l16][32 + lhalf * 16];
#pragma unroll
        for (int ch = 0; ch < 16; ++ch) {
            v16h nv0, nv1;
            if (ch < 15) {
                nv0 = *(const v16h*)&sVt[16 * (ch + 1) + l16][lhalf * 16];
                nv1 = *(const v16h*)&sVt[16 * (ch + 1) + l16][32 + lhalf * 16];
            }
            acc[ch] = __builtin_amdgcn_wmma_f32_16x16x32_f16(false, ap0, false, cv0,
                                                             (short)0, acc[ch], false, false);
            acc[ch] = __builtin_amdgcn_wmma_f32_16x16x32_f16(false, ap1, false, cv1,
                                                             (short)0, acc[ch], false, false);
            if (ch < 15) { cv0 = nv0; cv1 = nv1; }
        }
    }

    // ---- Normalize and store (fp32 out)
#pragma unroll
    for (int ch = 0; ch < 16; ++ch) {
#pragma unroll
        for (int r = 0; r < 8; ++r) {
            const int m = r + 8 * lhalf;
            O[(size_t)(qbase + m) * DIM + 16 * ch + l16] = acc[ch][r] / lrow[r];
        }
    }
}

extern "C" void kernel_launch(void* const* d_in, const int* in_sizes, int n_in,
                              void* d_out, int out_size, void* d_ws, size_t ws_size,
                              hipStream_t stream) {
    (void)in_sizes; (void)n_in; (void)out_size; (void)d_ws; (void)ws_size;
    const float* Q = (const float*)d_in[0];
    const float* K = (const float*)d_in[1];
    const float* V = (const float*)d_in[2];
    float* O = (float*)d_out;
    dim3 grid(SEQ / QROWS_PER_BLOCK);  // 128 blocks x 4 waves, 16 q-rows per wave
    dim3 block(BLOCK);
    fa_wmma_kernel<<<grid, block, 0, stream>>>(Q, K, V, O);
}
